// WITRAN_2DPSGMU_Encoder_24034636988655
// MI455X (gfx1250) — matrix operand order repaired and checked by reading, not compiled, not run on hardware
//
#include <hip/hip_runtime.h>
#include <hip/hip_bf16.h>

typedef __attribute__((ext_vector_type(16))) _Float16 v16h;
typedef __attribute__((ext_vector_type(8)))  float    v8f;

#define BATCH 32
#define SN    96      // slice_num (ROW)
#define OL    48      // original slice len (COL)
#define DIN   32
#define H     128
#define SL    143     // SN + OL - 1
#define NR    3072    // SN * BATCH
#define MB    16      // rows per workgroup
#define NWG   192     // NR / MB
#define K1    160     // DIN + H
#define KPRE  288     // 2H + DIN
#define NGATE 768     // 6H
#define A1S   168     // padded LDS row stride (halfs) for stage-1 A
#define APS   296     // padded LDS row stride (halfs) for A_pre
#define GS    776     // padded LDS row stride (halfs) for gate

union FragU { uint4 q[2]; v16h h; };

// A fragment, 16x32 f16 per ISA 7.12.2: lane m=lane&15; lanes<16 hold
// K in {8v.. } split pattern -> two contiguous 16B chunks at half*8 and 16+half*8.
__device__ inline v16h ld_frag_a(const _Float16* base, int strideH, int kt, int lane) {
  int m = lane & 15, hb = lane >> 4;
  const _Float16* p = base + m * strideH + kt * 32 + hb * 8;
  FragU u;
  u.q[0] = *(const uint4*)p;        // K = kt*32 + hb*8 .. +7
  u.q[1] = *(const uint4*)(p + 16); // K = kt*32 + 16 + hb*8 .. +7
  return u.h;
}

// B fragment, 32x16 f16: lane n=lane&15, lanes<16 hold K 0..7,16..23, lanes>=16 hold K 8..15,24..31.
// Weights stored row-major [n][k] so this is two 8-half runs at k and k+16 (2x b128).
__device__ inline v16h ld_frag_b(const _Float16* W, int strideH, int ntile, int kt, int lane) {
  int n = ntile * 16 + (lane & 15);
  int k = kt * 32 + (lane >> 4) * 8;
  const _Float16* p = W + (size_t)n * strideH + k;
  FragU u;
  u.q[0] = *(const uint4*)p;
  u.q[1] = *(const uint4*)(p + 16);
  return u.h;
}

__global__ __launch_bounds__(256) void witran_step(
    const float* __restrict__ input, const float* __restrict__ bf,
    const _Float16* __restrict__ W1r, const _Float16* __restrict__ W1c,
    const _Float16* __restrict__ Wf16,
    const float* __restrict__ b1r, const float* __restrict__ b1c,
    const float* __restrict__ hrowIn, float* __restrict__ hrowOut,
    const float* __restrict__ hcolIn, float* __restrict__ hcolOut,
    float* __restrict__ out, int t)
{
  __shared__ _Float16 lA1[2][MB * A1S];   // [0]=[xt|h_row], [1]=[xt|h_col]
  __shared__ _Float16 lApre[MB * APS];    // [rowm | colm | xt]
  __shared__ _Float16 lGate[MB * GS];     // gate pre-activations

  const int tid  = threadIdx.x;
  const int wave = tid >> 5, lane = tid & 31;
  const int g0 = blockIdx.x * MB;          // first global row of this tile
  const int r  = blockIdx.x >> 1;          // slice id (uniform in tile)
  const int bb = (blockIdx.x & 1) * 16;    // batch offset of this half-slice
  const int c  = t - r;                    // staggered column
  const bool valid = (c >= 0) && (c < OL);

  // ---------- load phase ----------
  { // x_t tile [16 x 32] -> f16 into both A1 buffers and A_pre cols 256..287
    int e = tid * 2;
#pragma unroll
    for (int i = 0; i < 2; ++i, ++e) {
      int m = e >> 5, k = e & 31;
      float v = 0.f;
      if (valid) {
        int bg = bb + m;
        v = input[(((size_t)bg * SN + r) * OL + c) * DIN + k];
      }
      _Float16 hv = (_Float16)v;
      lA1[0][m * A1S + k] = hv;
      lA1[1][m * A1S + k] = hv;
      lApre[m * APS + 256 + k] = hv;
    }
  }
  { // h_row / h_col tiles [16 x 128] f32 -> f16
    int m = tid >> 4, c0 = (tid & 15) * 8;
    const float* hr = hrowIn + (size_t)(g0 + m) * H + c0;
    const float* hc = hcolIn + (size_t)(g0 + m) * H + c0;
#pragma unroll
    for (int i = 0; i < 8; ++i) {
      lA1[0][m * A1S + 32 + c0 + i] = (_Float16)hr[i];
      lA1[1][m * A1S + 32 + c0 + i] = (_Float16)hc[i];
    }
  }
  __syncthreads();

  // ---------- stage 1: rowm/colm = [xt|h] @ [Wrxm|Wrm]^T, K=160 ----------
  {
    int ntile = wave;                     // 8 waves -> 8 N-tiles each state
    v8f acc0 = {}; v8f acc1 = {};
#pragma unroll
    for (int kt = 0; kt < 5; ++kt) {
      v16h a0 = ld_frag_a(lA1[0], A1S, kt, lane);
      v16h a1 = ld_frag_a(lA1[1], A1S, kt, lane);
      v16h b0 = ld_frag_b(W1r, K1, ntile, kt, lane);
      v16h b1 = ld_frag_b(W1c, K1, ntile, kt, lane);
      acc0 = __builtin_amdgcn_wmma_f32_16x16x32_f16(false, a0, false, b0,
                                                    (short)0, acc0, false, false);
      acc1 = __builtin_amdgcn_wmma_f32_16x16x32_f16(false, a1, false, b1,
                                                    (short)0, acc1, false, false);
    }
    int nl = lane & 15;
    int mb = (lane >> 4) * 8;             // C/D layout: m = v + 8*(lane>=16)
    float bi0 = b1r[ntile * 16 + nl];
    float bi1 = b1c[ntile * 16 + nl];
#pragma unroll
    for (int v = 0; v < 8; ++v) {
      int m = mb + v;
      lApre[m * APS + ntile * 16 + nl]       = (_Float16)(acc0[v] + bi0);
      lApre[m * APS + 128 + ntile * 16 + nl] = (_Float16)(acc1[v] + bi1);
    }
  }
  __syncthreads();

  // ---------- stage 2: gate = A_pre @ Wf^T, K=288, N=768 ----------
  {
    v8f acc[6] = {};
#pragma unroll
    for (int kt = 0; kt < 9; ++kt) {
      v16h a = ld_frag_a(lApre, APS, kt, lane);
#pragma unroll
      for (int j = 0; j < 6; ++j) {
        v16h b = ld_frag_b(Wf16, KPRE, wave * 6 + j, kt, lane);
        acc[j] = __builtin_amdgcn_wmma_f32_16x16x32_f16(false, a, false, b,
                                                        (short)0, acc[j], false, false);
      }
    }
    int nl = lane & 15;
    int mb = (lane >> 4) * 8;
#pragma unroll
    for (int j = 0; j < 6; ++j) {
      int n = (wave * 6 + j) * 16 + nl;
      float bi = bf[n];
#pragma unroll
      for (int v = 0; v < 8; ++v)
        lGate[(mb + v) * GS + n] = (_Float16)(acc[j][v] + bi);
    }
  }
  __syncthreads();

  // ---------- elementwise PSGMU update ----------
  {
    int m = tid >> 4, c0 = (tid & 15) * 8;
    int nrow = g0 + m;
    int bg   = bb + m;
    int rollRow = ((r + 1) % SN) * BATCH + bg;   // jnp.roll(h_col, B) applied at write
    const float* hrp = hrowIn + (size_t)nrow * H;
    const float* hcp = hcolIn + (size_t)nrow * H;
    float* orow   = out + ((size_t)nrow * SL + t) * 256;
    float* hidRow = out + 112459776ull;                 // NR*SL*2H
    float* hidCol = hidRow + 393216ull;                 // + B*SN*H
#pragma unroll
    for (int i = 0; i < 8; ++i) {
      int cc = c0 + i;
      float g_ur = (float)lGate[m * GS + cc];
      float g_or = (float)lGate[m * GS + 128 + cc];
      float g_uc = (float)lGate[m * GS + 256 + cc];
      float g_oc = (float)lGate[m * GS + 384 + cc];
      float g_ir = (float)lGate[m * GS + 512 + cc];
      float g_ic = (float)lGate[m * GS + 640 + cc];
      float ur  = 1.f / (1.f + __expf(-g_ur));
      float orr = 1.f / (1.f + __expf(-g_or));
      float uc  = 1.f / (1.f + __expf(-g_uc));
      float oc  = 1.f / (1.f + __expf(-g_oc));
      float ir  = tanhf(g_ir);
      float ic  = tanhf(g_ic);
      float hrv = tanhf((1.f - ur) * hrp[cc] + ur * ir) * orr;
      float hcv = tanhf((1.f - uc) * hcp[cc] + uc * ic) * oc;
      hrowOut[(size_t)nrow * H + cc]    = hrv;
      hcolOut[(size_t)rollRow * H + cc] = hcv;
      orow[cc]       = hrv;
      orow[128 + cc] = hcv;
      if (r == t - (OL - 1))
        hidRow[((size_t)bg * SN + r) * H + cc] = hrv;            // diagonal pick
      if (r == SN - 1 && t >= SN - 1)
        hidCol[((size_t)bg * OL + (t - (SN - 1))) * H + cc] = hcv; // tail pick
    }
  }
}

// Pack weights into f16, B-operand friendly row-major [n][k]; fuse biases.
__global__ void witran_prep(const float* __restrict__ Wf,
                            const float* __restrict__ Wrm, const float* __restrict__ brm,
                            const float* __restrict__ Wcm, const float* __restrict__ bcm,
                            const float* __restrict__ Wrxm, const float* __restrict__ brxm,
                            const float* __restrict__ Wcxm, const float* __restrict__ bcxm,
                            _Float16* __restrict__ W1r, _Float16* __restrict__ W1c,
                            _Float16* __restrict__ Wf16,
                            float* __restrict__ b1r, float* __restrict__ b1c)
{
  int i = blockIdx.x * blockDim.x + threadIdx.x;
  int stride = gridDim.x * blockDim.x;
  for (int idx = i; idx < NGATE * KPRE; idx += stride)
    Wf16[idx] = (_Float16)Wf[idx];
  for (int idx = i; idx < H * K1; idx += stride) {
    int n = idx / K1, k = idx % K1;
    float vr = (k < DIN) ? Wrxm[n * DIN + k] : Wrm[n * H + (k - DIN)];
    float vc = (k < DIN) ? Wcxm[n * DIN + k] : Wcm[n * H + (k - DIN)];
    W1r[idx] = (_Float16)vr;
    W1c[idx] = (_Float16)vc;
  }
  if (i < H) { b1r[i] = brxm[i] + brm[i]; b1c[i] = bcxm[i] + bcm[i]; }
}

__global__ void witran_zero(float* __restrict__ p, int n) {
  int i = blockIdx.x * blockDim.x + threadIdx.x;
  if (i < n) p[i] = 0.f;
}

extern "C" void kernel_launch(void* const* d_in, const int* in_sizes, int n_in,
                              void* d_out, int out_size, void* d_ws, size_t ws_size,
                              hipStream_t stream)
{
  (void)in_sizes; (void)n_in; (void)out_size; (void)ws_size;
  const float* input = (const float*)d_in[0];
  const float* Wf    = (const float*)d_in[1];
  const float* bf    = (const float*)d_in[2];
  const float* Wrm   = (const float*)d_in[3];
  const float* brm   = (const float*)d_in[4];
  const float* Wcm   = (const float*)d_in[5];
  const float* bcm   = (const float*)d_in[6];
  const float* Wrxm  = (const float*)d_in[7];
  const float* brxm  = (const float*)d_in[8];
  const float* Wcxm  = (const float*)d_in[9];
  const float* bcxm  = (const float*)d_in[10];

  // ws layout (bytes):
  //   0        h_row buf0   f32 [3072x128]
  //   1572864  h_row buf1
  //   3145728  h_col buf0
  //   4718592  h_col buf1
  //   6291456  W1r f16 [128x160]
  //   6332416  W1c f16 [128x160]
  //   6373376  Wf16 f16 [768x288]
  //   6815744  b1r f32 [128]
  //   6816256  b1c f32 [128]
  char* ws = (char*)d_ws;
  float*    hrow0 = (float*)(ws);
  float*    hrow1 = (float*)(ws + 1572864);
  float*    hcol0 = (float*)(ws + 3145728);
  float*    hcol1 = (float*)(ws + 4718592);
  _Float16* W1r   = (_Float16*)(ws + 6291456);
  _Float16* W1c   = (_Float16*)(ws + 6332416);
  _Float16* Wf16  = (_Float16*)(ws + 6373376);
  float*    b1r   = (float*)(ws + 6815744);
  float*    b1c   = (float*)(ws + 6816256);

  // zero h0 (all four ping-pong buffers)
  witran_zero<<<(1572864 + 255) / 256, 256, 0, stream>>>((float*)ws, 1572864);
  // pack weights (deterministic, graph-safe)
  witran_prep<<<512, 256, 0, stream>>>(Wf, Wrm, brm, Wcm, bcm, Wrxm, brxm, Wcxm, bcxm,
                                       W1r, W1c, Wf16, b1r, b1c);

  float* out = (float*)d_out;
  for (int t = 0; t < SL; ++t) {
    const float* hri = (t & 1) ? hrow1 : hrow0;
    float*       hro = (t & 1) ? hrow0 : hrow1;
    const float* hci = (t & 1) ? hcol1 : hcol0;
    float*       hco = (t & 1) ? hcol0 : hcol1;
    witran_step<<<NWG, 256, 0, stream>>>(input, bf, W1r, W1c, Wf16, b1r, b1c,
                                         hri, hro, hci, hco, out, t);
  }
}
